// LSTMFromScratch_33457795235963
// MI455X (gfx1250) — matrix operand order repaired and checked by reading, not compiled, not run on hardware
//
#include <hip/hip_runtime.h>

typedef __attribute__((ext_vector_type(8)))  __bf16 v8bf;
typedef __attribute__((ext_vector_type(16))) __bf16 v16bf;
typedef __attribute__((ext_vector_type(8)))  float  v8f;

#define B_  256
#define T_  1024
#define I_  64
#define H_  128
#define O_  64
#define KA_ 192                         /* H_ + I_ : combined [h | x] K dim */
#define GATE_W_ELEMS (4 * H_ * KA_)     /* 98304 bf16 */
#define WHY_ELEMS    (O_ * H_)          /* 8192 bf16  */
#define W_TOTAL      (GATE_W_ELEMS + WHY_ELEMS)

__device__ __forceinline__ float sigm(float v) {
  return 1.0f / (1.0f + __expf(-v));
}
__device__ __forceinline__ float tanh_f(float v) {
  float e = __expf(2.0f * v);
  return 1.0f - 2.0f / (e + 1.0f);   // safe at +/- inf
}

// ---------------------------------------------------------------------------
// Prep: convert weights fp32 -> bf16, transpose to [N][K] (K contiguous) so a
// WMMA B-fragment is one contiguous 32-byte LDS read per lane.
// Gate order in workspace: [f, i, c, o]; rows 0..127 = h-part, 128..191 = x-part
// (exactly the reference's W layout: W[:H] recurrent, W[H:] input).
// ---------------------------------------------------------------------------
__global__ void lstm_prep_kernel(const float* __restrict__ Wf,
                                 const float* __restrict__ Wi,
                                 const float* __restrict__ Wc,
                                 const float* __restrict__ Wo,
                                 const float* __restrict__ Why,
                                 __bf16* __restrict__ wsW,
                                 __bf16* __restrict__ wsWhy) {
  int idx = blockIdx.x * blockDim.x + threadIdx.x;
  if (idx < GATE_W_ELEMS) {
    int g = idx / (H_ * KA_);
    int r = idx % (H_ * KA_);
    int n = r / KA_;
    int k = r % KA_;
    const float* src = (g == 0) ? Wf : (g == 1) ? Wi : (g == 2) ? Wc : Wo;
    wsW[idx] = (__bf16)src[k * H_ + n];
  }
  if (idx < WHY_ELEMS) {
    int n = idx / H_;
    int k = idx % H_;
    wsWhy[idx] = (__bf16)Why[k * O_ + n];
  }
}

// ---------------------------------------------------------------------------
// Main persistent scan kernel: 16 workgroups x 256 threads (8 waves).
// Each WG owns a 16-row batch tile for all T steps; all weights LDS-resident.
// ---------------------------------------------------------------------------
__global__ __launch_bounds__(256) void lstm_scan_kernel(
    const float* __restrict__ x,   const float* __restrict__ h0,
    const float* __restrict__ c0,
    const float* __restrict__ bfp, const float* __restrict__ bip,
    const float* __restrict__ bop, const float* __restrict__ bcp,
    const float* __restrict__ byp,
    const __bf16* __restrict__ wsW,        // gate weights + Why, contiguous
    float* __restrict__ out) {

  __shared__ __align__(32) __bf16 sW[W_TOTAL];   // 212992 B
  __shared__ __align__(32) __bf16 sA[16][KA_];   // [h(0..127) | x(128..191)] bf16
  __shared__ float sG[4][16][H_];                // gate pre-activations
  __shared__ float sC[16][H_];                   // cell state (f32)

  const int tid  = threadIdx.x;
  const int lane = tid & 31;
  const int wave = tid >> 5;
  const int laneM  = lane & 15;   // A row / B,D column
  const int laneHi = lane >> 4;   // 0/1 half-wave
  const int b0 = blockIdx.x * 16;

  // ---- cache all weights in LDS (one time) ----
  {
    const uint4* src = (const uint4*)wsW;
    uint4*       dst = (uint4*)sW;
    const int n16 = W_TOTAL * 2 / 16;            // 13312 x 16B
    for (int i = tid; i < n16; i += 256) dst[i] = src[i];
  }

  // ---- init h (bf16 into A panel), c, x_0 ----
  for (int i = tid; i < 16 * H_; i += 256) {
    int m = i / H_, n = i % H_;
    sA[m][n] = (__bf16)h0[(b0 + m) * H_ + n];
    sC[m][n] = c0[(b0 + m) * H_ + n];
  }
  for (int i = tid; i < 16 * I_; i += 256) {
    int m = i >> 6, k = i & 63;
    sA[m][H_ + k] = (__bf16)x[((size_t)(b0 + m) * T_) * I_ + k];
  }

  // ---- per-thread elementwise ownership: (m, n0..n0+7); biases in regs ----
  const int em  = tid & 15;
  const int en0 = (tid >> 4) * 8;
  float rbf[8], rbi[8], rbc[8], rbo[8];
#pragma unroll
  for (int j = 0; j < 8; ++j) {
    rbf[j] = bfp[en0 + j]; rbi[j] = bip[en0 + j];
    rbc[j] = bcp[en0 + j]; rbo[j] = bop[en0 + j];
  }

  // wave tile assignment for gate matmul: gate g = wave/2, 4 col-tiles each
  const int g      = wave >> 1;
  const int ntBase = (wave & 1) * 4;
  const __bf16* wg = sW + g * (H_ * KA_);
  const __bf16* wy = sW + GATE_W_ELEMS;
  const float byv  = byp[wave < 4 ? (wave * 16 + laneM) : 0];

  __syncthreads();

  for (int t = 0; t < T_; ++t) {
    // ---------------- gate matmul: [h|x](16x192) @ W (192x128) x 4 gates ----
    v8f acc[4] = {v8f{}, v8f{}, v8f{}, v8f{}};
#pragma unroll
    for (int kk = 0; kk < 6; ++kk) {
      const int kbase = kk * 32 + laneHi * 16;
      v8bf a0 = *(const v8bf*)&sA[laneM][kbase];
      v8bf a1 = *(const v8bf*)&sA[laneM][kbase + 8];
      v16bf Av = __builtin_shufflevector(a0, a1, 0, 1, 2, 3, 4, 5, 6, 7,
                                         8, 9, 10, 11, 12, 13, 14, 15);
#pragma unroll
      for (int i = 0; i < 4; ++i) {
        const int N = (ntBase + i) * 16 + laneM;
        v16bf Bv = *(const v16bf*)&wg[N * KA_ + kk * 32 + laneHi * 16];
        acc[i] = __builtin_amdgcn_wmma_f32_16x16x32_bf16(
            false, Av, false, Bv, (short)0, acc[i], false, false);
      }
    }
#pragma unroll
    for (int i = 0; i < 4; ++i) {
      const int N = (ntBase + i) * 16 + laneM;
#pragma unroll
      for (int r = 0; r < 8; ++r) sG[g][r + 8 * laneHi][N] = acc[i][r];
    }
    __syncthreads();

    // ---------------- elementwise gate fusion + state update ----------------
#pragma unroll
    for (int j = 0; j < 8; ++j) {
      const int n = en0 + j;
      float ft = sigm(sG[0][em][n] + rbf[j]);
      float it = sigm(sG[1][em][n] + rbi[j]);
      float ch = tanh_f(sG[2][em][n] + rbc[j]);
      float ot = sigm(sG[3][em][n] + rbo[j]);
      float c  = ft * sC[em][n] + it * ch;
      sC[em][n] = c;
      float h = ot * tanh_f(c);
      sA[em][n] = (__bf16)h;
      if (t == T_ - 1) {
        out[(size_t)B_ * T_ * O_ + (size_t)(b0 + em) * H_ + n] = h;
        out[(size_t)B_ * T_ * O_ + (size_t)B_ * H_ +
            (size_t)(b0 + em) * H_ + n] = c;
      }
    }
    // prefetch + convert x_{t+1} into the A panel
    if (t + 1 < T_) {
      for (int i = tid; i < 16 * I_; i += 256) {
        int m = i >> 6, k = i & 63;
        sA[m][H_ + k] =
            (__bf16)x[((size_t)(b0 + m) * T_ + (t + 1)) * I_ + k];
      }
    }
    __syncthreads();

    // ---------------- y = h @ Why + by  (waves 0..3, one 16-col tile each) --
    if (wave < 4) {
      v8f yacc = {};
      const int N = wave * 16 + laneM;
#pragma unroll
      for (int kk = 0; kk < 4; ++kk) {
        const int kbase = kk * 32 + laneHi * 16;
        v8bf a0 = *(const v8bf*)&sA[laneM][kbase];
        v8bf a1 = *(const v8bf*)&sA[laneM][kbase + 8];
        v16bf Av = __builtin_shufflevector(a0, a1, 0, 1, 2, 3, 4, 5, 6, 7,
                                           8, 9, 10, 11, 12, 13, 14, 15);
        v16bf Bv = *(const v16bf*)&wy[N * H_ + kk * 32 + laneHi * 16];
        yacc = __builtin_amdgcn_wmma_f32_16x16x32_bf16(
            false, Av, false, Bv, (short)0, yacc, false, false);
      }
#pragma unroll
      for (int r = 0; r < 8; ++r) {
        const int M = r + 8 * laneHi;
        out[((size_t)(b0 + M) * T_ + t) * O_ + N] = yacc[r] + byv;
      }
    }
    __syncthreads();
  }
}

// ---------------------------------------------------------------------------
extern "C" void kernel_launch(void* const* d_in, const int* in_sizes, int n_in,
                              void* d_out, int out_size, void* d_ws,
                              size_t ws_size, hipStream_t stream) {
  const float* x   = (const float*)d_in[0];
  const float* h0  = (const float*)d_in[1];
  const float* c0  = (const float*)d_in[2];
  const float* Wf  = (const float*)d_in[3];
  const float* Wi  = (const float*)d_in[4];
  const float* Wo  = (const float*)d_in[5];
  const float* Wc  = (const float*)d_in[6];
  const float* bf  = (const float*)d_in[7];
  const float* bi  = (const float*)d_in[8];
  const float* bo  = (const float*)d_in[9];
  const float* bc  = (const float*)d_in[10];
  const float* Why = (const float*)d_in[11];
  const float* by  = (const float*)d_in[12];
  float* out = (float*)d_out;

  __bf16* wsW   = (__bf16*)d_ws;
  __bf16* wsWhy = wsW + GATE_W_ELEMS;

  lstm_prep_kernel<<<(GATE_W_ELEMS + 255) / 256, 256, 0, stream>>>(
      Wf, Wi, Wc, Wo, Why, wsW, wsWhy);
  lstm_scan_kernel<<<B_ / 16, 256, 0, stream>>>(
      x, h0, c0, bf, bi, bo, bc, by, wsW, out);
}
